// CausalSelfAttention_26568667693077
// MI455X (gfx1250) — matrix operand order repaired and checked by reading, not compiled, not run on hardware
//
#include <hip/hip_runtime.h>
#include <hip/hip_bf16.h>

typedef __attribute__((ext_vector_type(16))) __bf16 v16bf;
typedef __attribute__((ext_vector_type(8)))  __bf16 v8bf;
typedef __attribute__((ext_vector_type(4)))  __bf16 v4bf;
typedef __attribute__((ext_vector_type(8)))  float  v8f;
typedef __attribute__((ext_vector_type(4)))  float  v4f;

#define NB 2
#define NT 2048
#define NC 1024
#define NH 16
#define ND 64
#define MTOT (NB * NT)     // 4096
#define NQKV (3 * NC)      // 3072

// ---- CDNA5 async global->LDS (ASYNCcnt) with safe fallback --------------
#if defined(__HIP_DEVICE_COMPILE__) && __has_builtin(__builtin_amdgcn_global_load_async_to_lds_b128)
#define USE_ASYNC_LDS 1
#endif

#ifdef USE_ASYNC_LDS
// builtin signature: (int4 AS1* src, int4 AS3* dst, imm offset, imm cpol)
typedef int v4i_vs __attribute__((vector_size(16)));
typedef __attribute__((address_space(1))) v4i_vs gv4i;
typedef __attribute__((address_space(3))) v4i_vs lv4i;
#endif

__device__ __forceinline__ void g2l_b128(const __bf16* gsrc, __bf16* ldst) {
#ifdef USE_ASYNC_LDS
    __builtin_amdgcn_global_load_async_to_lds_b128((gv4i*)gsrc, (lv4i*)ldst, 0, 0);
#else
    *(v8bf*)ldst = *(const v8bf*)gsrc;
#endif
}
__device__ __forceinline__ void async_wait() {
#ifdef USE_ASYNC_LDS
#if __has_builtin(__builtin_amdgcn_s_wait_asynccnt)
    __builtin_amdgcn_s_wait_asynccnt(0);
#else
    asm volatile("s_wait_asynccnt 0x0" ::: "memory");
#endif
#endif
}

// ---- WMMA fragment layouts (CDNA5 ISA 7.12.2, wave32, 16-bit data) ------
// A 16x32: lane=(row&15)|((k>>3)&1)<<4 ; elem=(k&7)|((k>>4)&1)<<3
// B 32x16 (k x n): lane=(n&15)|((k>>3)&1)<<4 ; elem=(k&7)|((k>>4)&1)<<3
// LDS holds fragments "ready": offset = lane*16 + elem  (v16bf = 1 lane read)
__device__ __forceinline__ int a_frag_off(int row, int k) {
    return (((row & 15) | (((k >> 3) & 1) << 4)) << 4) | ((k & 7) | (((k >> 4) & 1) << 3));
}
__device__ __forceinline__ int b_frag_off(int k, int n) {
    return (((n & 15) | (((k >> 3) & 1) << 4)) << 4) | ((k & 7) | (((k >> 4) & 1) << 3));
}
// C/D 16x16 f32: lane holds col n=lane&15; VGPR j holds row j+(lane>>4)*8.

// =====================================================================
// Kernel 1: qkv = x @ Wqkv + bqkv -> bf16 [3][B][H][T][D], Q pre-scaled.
// 256 thr (8 waves), tile 128x64, K-step 32, double-buffered LDS.
// =====================================================================
__global__ __launch_bounds__(256) void qkv_gemm_kernel(
    const float* __restrict__ x, const float* __restrict__ W,
    const float* __restrict__ bias, __bf16* __restrict__ qkv)
{
    __shared__ __align__(32) __bf16 sA[2][8 * 512];  // 128x32 frags
    __shared__ __align__(32) __bf16 sB[2][4 * 512];  // 32x64  frags

    const int tid  = threadIdx.x;
    const int lane = tid & 31;
    const int w    = tid >> 5;
    const int wm   = w >> 1, wn = w & 1;     // 4(M) x 2(N) waves
    const int m0   = blockIdx.x * 128;
    const int n0   = blockIdx.y * 64;

    const float* xb0 = x + (size_t)m0 * NC;          // uniform bases
    const float* Wb0 = W + n0;

    auto stage = [&](int kk, int p) {
        const float* xb = xb0 + kk;                   // uniform
        const float* Wb = Wb0 + (size_t)kk * NQKV;    // uniform
        #pragma unroll
        for (int i = 0; i < 4; ++i) {                 // A: 1024 float4
            int e = tid + (i << 8);
            int r = e >> 3, c4 = (e & 7) << 2;
            v4f v = *(const v4f*)(xb + (unsigned)(r * NC + c4));
            v4bf bv;
            #pragma unroll
            for (int t = 0; t < 4; ++t) bv[t] = (__bf16)v[t];
            *(v4bf*)&sA[p][((r >> 4) << 9) + a_frag_off(r & 15, c4)] = bv;
        }
        #pragma unroll
        for (int i = 0; i < 2; ++i) {                 // B: 512 float4
            int e = tid + (i << 8);
            int r = e >> 4, c4 = (e & 15) << 2;
            v4f v = *(const v4f*)(Wb + (unsigned)(r * NQKV + c4));
            #pragma unroll
            for (int t = 0; t < 4; ++t) {
                int ct = c4 + t;
                sB[p][((ct >> 4) << 9) + b_frag_off(r, ct & 15)] = (__bf16)v[t];
            }
        }
    };

    v8f acc[2][2];
    #pragma unroll
    for (int mi = 0; mi < 2; ++mi)
        #pragma unroll
        for (int ni = 0; ni < 2; ++ni)
            #pragma unroll
            for (int j = 0; j < 8; ++j) acc[mi][ni][j] = 0.0f;

    stage(0, 0);
    async_wait();
    __syncthreads();

    for (int kk = 0; kk < NC; kk += 32) {
        const int p = (kk >> 5) & 1;
        if (kk + 32 < NC) stage(kk + 32, p ^ 1);      // overlap fetch w/ WMMA

        v16bf af[2], bf[2];
        #pragma unroll
        for (int mi = 0; mi < 2; ++mi)
            af[mi] = *(const v16bf*)&sA[p][(((wm << 1) + mi) << 9) + (lane << 4)];
        #pragma unroll
        for (int ni = 0; ni < 2; ++ni)
            bf[ni] = *(const v16bf*)&sB[p][(((wn << 1) + ni) << 9) + (lane << 4)];
        #pragma unroll
        for (int mi = 0; mi < 2; ++mi)
            #pragma unroll
            for (int ni = 0; ni < 2; ++ni)
                acc[mi][ni] = __builtin_amdgcn_wmma_f32_16x16x32_bf16(
                    false, af[mi], false, bf[ni], (short)0, acc[mi][ni], false, false);

        async_wait();
        __syncthreads();
    }

    // epilogue: bias, scale Q by 1/8, scatter to [3][B][H][T][D] bf16
    #pragma unroll
    for (int mi = 0; mi < 2; ++mi)
        #pragma unroll
        for (int ni = 0; ni < 2; ++ni) {
            int ng = n0 + (wn << 5) + (ni << 4) + (lane & 15);
            float bv = bias[ng];
            int which = ng >> 10, cc = ng & 1023;
            int h = cc >> 6, d = cc & 63;
            float scale = (which == 0) ? 0.125f : 1.0f;
            #pragma unroll
            for (int j = 0; j < 8; ++j) {
                int mg = m0 + (wm << 5) + (mi << 4) + j + ((lane >> 4) << 3);
                int b = mg >> 11, t = mg & 2047;
                float v = (acc[mi][ni][j] + bv) * scale;
                unsigned off = ((((unsigned)which * NB + b) * NH + h) * NT + t) * ND + d;
                qkv[off] = (__bf16)v;
            }
        }
}

// =====================================================================
// Kernel 2: flash attention per (b,h). 128 thr (4 waves), q-tile 64.
// =====================================================================
__global__ __launch_bounds__(128) void attn_kernel(
    const __bf16* __restrict__ qkv, __bf16* __restrict__ y)
{
    __shared__ __align__(32) __bf16 sQ[8 * 512];  // 64x64 A-frags (q x d)
    __shared__ __align__(32) __bf16 sK[8 * 512];  // 64x64 B-frags (d x key)
    __shared__ __align__(32) __bf16 sV[8 * 512];  // 64x64 B-frags (key x d)
    __shared__ __align__(32) __bf16 sP[8 * 512];  // 64x64 A-frags (q x key)

    const int tid  = threadIdx.x;
    const int lane = tid & 31;
    const int w    = tid >> 5;
    const int qt   = blockIdx.x;
    const int bh   = blockIdx.y;
    const int b = bh >> 4, h = bh & 15;
    const int q0 = qt * 64;

    const size_t head_off = ((size_t)(b * NH + h)) * NT * ND;
    const __bf16* qb = qkv + head_off + (size_t)q0 * ND;
    const __bf16* kbase = qkv + (size_t)NB * NH * NT * ND + head_off;
    const __bf16* vbase = qkv + (size_t)2 * NB * NH * NT * ND + head_off;

    // stage Q via async DMA (frag-ready layout, b128 per lane)
    #pragma unroll
    for (int i = 0; i < 4; ++i) {
        int e = tid + (i << 7);
        int r = e >> 3, c8 = (e & 7) << 3;
        g2l_b128(qb + (unsigned)(r * ND + c8),
                 &sQ[((((r >> 4) << 1) + (c8 >> 5)) << 9) + a_frag_off(r & 15, c8 & 31)]);
    }

    float m_run[8], l_run[8];
    v8f accO[4];
    #pragma unroll
    for (int j = 0; j < 8; ++j) { m_run[j] = -1e30f; l_run[j] = 0.0f; }
    #pragma unroll
    for (int dt = 0; dt < 4; ++dt)
        #pragma unroll
        for (int j = 0; j < 8; ++j) accO[dt][j] = 0.0f;

    async_wait();
    __syncthreads();

    for (int jt = 0; jt <= qt; ++jt) {              // causal tile skip
        const __bf16* kb = kbase + (size_t)(jt * 64) * ND;
        const __bf16* vb = vbase + (size_t)(jt * 64) * ND;
        #pragma unroll
        for (int i = 0; i < 4; ++i) {
            int e = tid + (i << 7);
            int r = e >> 3, c8 = (e & 7) << 3;
            unsigned goff = (unsigned)(r * ND + c8);
            g2l_b128(kb + goff,                      // K: contiguous frag chunk
                     &sK[((((r >> 4) << 1) + (c8 >> 5)) << 9) + b_frag_off(c8 & 31, r & 15)]);
            v8bf vv = *(const v8bf*)(vb + goff);     // V: needs transpose-scatter
            #pragma unroll
            for (int t = 0; t < 8; ++t) {
                int ct = c8 + t;
                sV[((((ct >> 4) << 1) + (r >> 5)) << 9) + b_frag_off(r & 31, ct & 15)] = vv[t];
            }
        }
        async_wait();
        __syncthreads();

        // S = Q K^T : 16 q-rows x 64 keys per wave
        v8f sacc[4];
        #pragma unroll
        for (int nt = 0; nt < 4; ++nt) {
            #pragma unroll
            for (int j = 0; j < 8; ++j) sacc[nt][j] = 0.0f;
            #pragma unroll
            for (int kg = 0; kg < 2; ++kg) {
                v16bf a  = *(const v16bf*)&sQ[(((w << 1) + kg) << 9) + (lane << 4)];
                v16bf bb = *(const v16bf*)&sK[(((nt << 1) + kg) << 9) + (lane << 4)];
                sacc[nt] = __builtin_amdgcn_wmma_f32_16x16x32_bf16(
                    false, a, false, bb, (short)0, sacc[nt], false, false);
            }
        }
        if (jt == qt) {                              // diagonal causal mask
            #pragma unroll
            for (int nt = 0; nt < 4; ++nt) {
                int kgidx = jt * 64 + (nt << 4) + (lane & 15);
                #pragma unroll
                for (int j = 0; j < 8; ++j) {
                    int qg = q0 + (w << 4) + j + ((lane >> 4) << 3);
                    if (kgidx > qg) sacc[nt][j] = -1e30f;
                }
            }
        }
        // online softmax (rows live in one lane-half; xor-shuffle 16 lanes)
        float mnew[8], alpha[8], lsum[8];
        #pragma unroll
        for (int j = 0; j < 8; ++j) {
            float mx = m_run[j];
            #pragma unroll
            for (int nt = 0; nt < 4; ++nt) mx = fmaxf(mx, sacc[nt][j]);
            #pragma unroll
            for (int off = 1; off < 16; off <<= 1)
                mx = fmaxf(mx, __shfl_xor(mx, off, 32));
            mnew[j] = mx;
            alpha[j] = __expf(m_run[j] - mx);
            lsum[j] = 0.0f;
        }
        #pragma unroll
        for (int nt = 0; nt < 4; ++nt)
            #pragma unroll
            for (int j = 0; j < 8; ++j) {
                float p = __expf(sacc[nt][j] - mnew[j]);
                sacc[nt][j] = p;
                lsum[j] += p;
            }
        #pragma unroll
        for (int j = 0; j < 8; ++j) {
            float s = lsum[j];
            #pragma unroll
            for (int off = 1; off < 16; off <<= 1)
                s += __shfl_xor(s, off, 32);
            l_run[j] = l_run[j] * alpha[j] + s;
            m_run[j] = mnew[j];
        }
        // P -> LDS A-frags (same-wave region; DS ops in-order, no barrier)
        #pragma unroll
        for (int nt = 0; nt < 4; ++nt) {
            int col = (nt << 4) + (lane & 15);
            #pragma unroll
            for (int j = 0; j < 8; ++j) {
                int row = j + ((lane >> 4) << 3);
                sP[(((w << 1) + (col >> 5)) << 9) + a_frag_off(row, col & 31)] =
                    (__bf16)sacc[nt][j];
            }
        }
        #pragma unroll
        for (int dt = 0; dt < 4; ++dt)
            #pragma unroll
            for (int j = 0; j < 8; ++j) accO[dt][j] *= alpha[j];
        #pragma unroll
        for (int dt = 0; dt < 4; ++dt)
            #pragma unroll
            for (int kg = 0; kg < 2; ++kg) {
                v16bf a  = *(const v16bf*)&sP[(((w << 1) + kg) << 9) + (lane << 4)];
                v16bf bb = *(const v16bf*)&sV[(((dt << 1) + kg) << 9) + (lane << 4)];
                accO[dt] = __builtin_amdgcn_wmma_f32_16x16x32_bf16(
                    false, a, false, bb, (short)0, accO[dt], false, false);
            }
        __syncthreads();
    }

    // y[b, t, h*64+d] = O / l   (bf16 operand for proj GEMM)
    #pragma unroll
    for (int dt = 0; dt < 4; ++dt) {
        int d = (dt << 4) + (lane & 15);
        #pragma unroll
        for (int j = 0; j < 8; ++j) {
            int t = q0 + (w << 4) + j + ((lane >> 4) << 3);
            float v = accO[dt][j] / l_run[j];
            unsigned off = ((unsigned)(b * NT + t)) * NC + (h << 6) + d;
            y[off] = (__bf16)v;
        }
    }
}

// =====================================================================
// Kernel 3: out = y @ Wproj + bproj (A bf16 via async DMA, out fp32)
// =====================================================================
__global__ __launch_bounds__(256) void proj_gemm_kernel(
    const __bf16* __restrict__ A, const float* __restrict__ W,
    const float* __restrict__ bias, float* __restrict__ out)
{
    __shared__ __align__(32) __bf16 sA[2][8 * 512];
    __shared__ __align__(32) __bf16 sB[2][4 * 512];

    const int tid  = threadIdx.x;
    const int lane = tid & 31;
    const int w    = tid >> 5;
    const int wm   = w >> 1, wn = w & 1;
    const int m0   = blockIdx.x * 128;
    const int n0   = blockIdx.y * 64;

    const __bf16* Ab0 = A + (size_t)m0 * NC;
    const float*  Wb0 = W + n0;

    auto stage = [&](int kk, int p) {
        const __bf16* Ab = Ab0 + kk;
        const float*  Wb = Wb0 + (size_t)kk * NC;
        #pragma unroll
        for (int i = 0; i < 2; ++i) {                // A: 512 x b128 async
            int e = tid + (i << 8);
            int r = e >> 2, c8 = (e & 3) << 3;
            g2l_b128(Ab + (unsigned)(r * NC + c8),
                     &sA[p][((r >> 4) << 9) + a_frag_off(r & 15, c8)]);
        }
        #pragma unroll
        for (int i = 0; i < 2; ++i) {                // B: 512 float4
            int e = tid + (i << 8);
            int r = e >> 4, c4 = (e & 15) << 2;
            v4f v = *(const v4f*)(Wb + (unsigned)(r * NC + c4));
            #pragma unroll
            for (int t = 0; t < 4; ++t) {
                int ct = c4 + t;
                sB[p][((ct >> 4) << 9) + b_frag_off(r, ct & 15)] = (__bf16)v[t];
            }
        }
    };

    v8f acc[2][2];
    #pragma unroll
    for (int mi = 0; mi < 2; ++mi)
        #pragma unroll
        for (int ni = 0; ni < 2; ++ni)
            #pragma unroll
            for (int j = 0; j < 8; ++j) acc[mi][ni][j] = 0.0f;

    stage(0, 0);
    async_wait();
    __syncthreads();

    for (int kk = 0; kk < NC; kk += 32) {
        const int p = (kk >> 5) & 1;
        if (kk + 32 < NC) stage(kk + 32, p ^ 1);

        v16bf af[2], bf[2];
        #pragma unroll
        for (int mi = 0; mi < 2; ++mi)
            af[mi] = *(const v16bf*)&sA[p][(((wm << 1) + mi) << 9) + (lane << 4)];
        #pragma unroll
        for (int ni = 0; ni < 2; ++ni)
            bf[ni] = *(const v16bf*)&sB[p][(((wn << 1) + ni) << 9) + (lane << 4)];
        #pragma unroll
        for (int mi = 0; mi < 2; ++mi)
            #pragma unroll
            for (int ni = 0; ni < 2; ++ni)
                acc[mi][ni] = __builtin_amdgcn_wmma_f32_16x16x32_bf16(
                    false, af[mi], false, bf[ni], (short)0, acc[mi][ni], false, false);

        async_wait();
        __syncthreads();
    }

    #pragma unroll
    for (int mi = 0; mi < 2; ++mi)
        #pragma unroll
        for (int ni = 0; ni < 2; ++ni) {
            int ng = n0 + (wn << 5) + (ni << 4) + (lane & 15);
            float bv = bias[ng];
            #pragma unroll
            for (int j = 0; j < 8; ++j) {
                int mg = m0 + (wm << 5) + (mi << 4) + j + ((lane >> 4) << 3);
                out[(unsigned)mg * NC + ng] = acc[mi][ni][j] + bv;
            }
        }
}

// =====================================================================
extern "C" void kernel_launch(void* const* d_in, const int* in_sizes, int n_in,
                              void* d_out, int out_size, void* d_ws, size_t ws_size,
                              hipStream_t stream) {
    const float* x     = (const float*)d_in[0];
    const float* Wqkv  = (const float*)d_in[1];
    const float* bqkv  = (const float*)d_in[2];
    const float* Wproj = (const float*)d_in[3];
    const float* bproj = (const float*)d_in[4];
    float* out = (float*)d_out;

    // ws: [0,24MB) qkv bf16 [3][B][H][T][D]; [24MB,32MB) y bf16 [B,T,C]
    __bf16* qkvbuf = (__bf16*)d_ws;
    __bf16* ybuf   = qkvbuf + (size_t)3 * NB * NT * NC;

    dim3 g1(MTOT / 128, NQKV / 64);           // 32 x 48
    hipLaunchKernelGGL(qkv_gemm_kernel, g1, dim3(256), 0, stream, x, Wqkv, bqkv, qkvbuf);

    dim3 g2(NT / 64, NB * NH);                // 32 x 32
    hipLaunchKernelGGL(attn_kernel, g2, dim3(128), 0, stream, qkvbuf, ybuf);

    dim3 g3(MTOT / 128, NC / 64);             // 32 x 16
    hipLaunchKernelGGL(proj_gemm_kernel, g3, dim3(256), 0, stream, ybuf, Wproj, bproj, out);
}
